// GraphConvLayer_48120813584763
// MI455X (gfx1250) — matrix operand order repaired and checked by reading, not compiled, not run on hardware
//
#include <hip/hip_runtime.h>
#include <hip/hip_bf16.h>

typedef __bf16 bf16_t;
typedef __attribute__((ext_vector_type(16))) __bf16 v16bf;
typedef __attribute__((ext_vector_type(8)))  float  v8f;

#define NNODES 10000
#define NEDGES 320000
#define DIM 256

static __device__ __forceinline__ void split_bf16(float f, bf16_t& hi, bf16_t& lo) {
  hi = (bf16_t)f;
  lo = (bf16_t)(f - (float)hi);
}

// Pre-swizzle a row-major fp32 weight matrix W[K x 256] into the ISA 32x16
// bf16 B-fragment layout (hi/lo split), so the GEMM hot loop does contiguous
// 16B vector loads. One thread per (kstep, ntile16, lane): writes 16 bf16.
// Fragment element i of lane L in tile (ks, nt) is W[kb + i + (i&8)][nt*16 + (L&15)]
// with kb = ks*32 + ((L&16) ? 8 : 0)   (per CDNA5 ISA B layout).
__global__ void __launch_bounds__(256)
swizzle_w_kernel(const float* __restrict__ W, int ksteps,
                 bf16_t* __restrict__ swhi, bf16_t* __restrict__ swlo) {
  int t = blockIdx.x * blockDim.x + threadIdx.x;
  int total = ksteps * 16 * 32;
  if (t >= total) return;
  int lane  = t & 31;
  int ntile = (t >> 5) & 15;
  int kstep = t >> 9;
  int n  = ntile * 16 + (lane & 15);
  int kb = kstep * 32 + ((lane & 16) ? 8 : 0);
  bf16_t* oh = swhi + (size_t)t * 16;
  bf16_t* ol = swlo + (size_t)t * 16;
#pragma unroll
  for (int i = 0; i < 16; ++i) {
    float f = W[(size_t)(kb + i + (i & 8)) * DIM + n];
    bf16_t h, l;
    split_bf16(f, h, l);
    oh[i] = h;
    ol[i] = l;
  }
}

__global__ void __launch_bounds__(256)
zero_kernel(float4* __restrict__ p, int n4) {
  int t = blockIdx.x * blockDim.x + threadIdx.x;
  if (t < n4) p[t] = make_float4(0.f, 0.f, 0.f, 0.f);
}

// C[M x 256] = relu?( [A0 | A1][M x Ktot] @ Wsw + bias )
// A0/A1 are row-major fp32 with row length 256 (A1 used for k >= 256).
// Each wave computes a 16x64 tile: 4 f32 accumulators, hi/lo bf16 split
// (3 x v_wmma_f32_16x16x32_bf16 per 16x16 subtile per 32-wide k-step).
// M = 10000 = 625*16 exactly, N = 256 exactly -> no edge masking, EXEC all-1s.
__global__ void __launch_bounds__(256)
gemm_wmma_kernel(const float* __restrict__ A0, const float* __restrict__ A1,
                 int Ktot,
                 const bf16_t* __restrict__ swhi, const bf16_t* __restrict__ swlo,
                 float* __restrict__ C, int ldc,
                 const float* __restrict__ bias, int doRelu) {
  const int wid  = threadIdx.x >> 5;
  const int lane = threadIdx.x & 31;
  int wt = blockIdx.x * 8 + wid;            // wave-uniform
  if (wt >= 625 * 4) return;
  const int mtile = wt >> 2;
  const int n64   = wt & 3;
  const int m0  = mtile * 16;
  const int nt0 = n64 * 4;                  // first 16-wide n-tile of this wave

  const v8f vzero = {0.f, 0.f, 0.f, 0.f, 0.f, 0.f, 0.f, 0.f};
  v8f acc[4];
#pragma unroll
  for (int s = 0; s < 4; ++s) acc[s] = vzero;

  // A-fragment addressing per 16-bit A layout (16x32):
  // lanes 0-15 hold K = k+0..7 and k+16..23 of row (m0 + lane),
  // lanes 16-31 hold K = k+8..15 and k+24..31 of row (m0 + lane-16).
  const int mrow = m0 + (lane & 15);
  const int koff = (lane & 16) ? 8 : 0;
  const int ksteps = Ktot >> 5;

  for (int ks = 0; ks < ksteps; ++ks) {
    const int kk = ks << 5;
    const float* Aseg;
    int kloc;
    if (kk < 256) { Aseg = A0; kloc = kk; } else { Aseg = A1; kloc = kk - 256; }
    const float* ap = Aseg + (size_t)mrow * DIM + kloc + koff;
    float4 f0 = *(const float4*)(ap + 0);
    float4 f1 = *(const float4*)(ap + 4);
    float4 f2 = *(const float4*)(ap + 16);
    float4 f3 = *(const float4*)(ap + 20);
    float av[16] = {f0.x, f0.y, f0.z, f0.w, f1.x, f1.y, f1.z, f1.w,
                    f2.x, f2.y, f2.z, f2.w, f3.x, f3.y, f3.z, f3.w};
    v16bf a_hi, a_lo;
#pragma unroll
    for (int i = 0; i < 16; ++i) {
      bf16_t h, l;
      split_bf16(av[i], h, l);
      a_hi[i] = h;
      a_lo[i] = l;
    }

    const size_t bstep = (size_t)ks * 16 * 32 * 16;  // elements per k-step block
#pragma unroll
    for (int s = 0; s < 4; ++s) {
      const size_t boff = bstep + ((size_t)(nt0 + s) * 32 + lane) * 16;
      v16bf b_hi = *(const v16bf*)(swhi + boff);
      v16bf b_lo = *(const v16bf*)(swlo + boff);
      acc[s] = __builtin_amdgcn_wmma_f32_16x16x32_bf16(
          false, a_hi, false, b_hi, (short)0, acc[s], false, false);
      acc[s] = __builtin_amdgcn_wmma_f32_16x16x32_bf16(
          false, a_hi, false, b_lo, (short)0, acc[s], false, false);
      acc[s] = __builtin_amdgcn_wmma_f32_16x16x32_bf16(
          false, a_lo, false, b_hi, (short)0, acc[s], false, false);
    }
  }

  // C/D 16x16 f32 layout: N = lane&15 (+subtile), VGPR r -> M = r (+8 for lanes>=16)
  const int nbase = n64 * 64 + (lane & 15);
  const int mbase = m0 + ((lane & 16) ? 8 : 0);
#pragma unroll
  for (int s = 0; s < 4; ++s) {
    const int n = nbase + s * 16;
    const float bv = bias ? bias[n] : 0.f;
#pragma unroll
    for (int r = 0; r < 8; ++r) {
      float v = acc[s][r] + bv;
      if (doRelu) v = fmaxf(v, 0.f);
      C[(size_t)(mbase + r) * ldc + n] = v;
    }
  }
}

// One wave per edge: msg = relu(P[row] + Q[col] + b_msg); atomicAdd into agg[col].
// PQ is [N, 512] with P in cols 0..255 and Q in cols 256..511.
__global__ void __launch_bounds__(256)
edge_kernel(const float* __restrict__ PQ, const int* __restrict__ ei,
            const float* __restrict__ bmsg, float* __restrict__ agg) {
  int t = blockIdx.x * blockDim.x + threadIdx.x;
  int e = t >> 5;                            // wave-uniform
  if (e >= NEDGES) return;
  int lane = t & 31;
  int row = ei[e];
  int col = ei[NEDGES + e];
  int j = lane * 8;
  const float4* P = (const float4*)(PQ + (size_t)row * 512 + j);
  const float4* Q = (const float4*)(PQ + (size_t)col * 512 + 256 + j);
  const float4* B = (const float4*)(bmsg + j);
  float4 p0 = P[0], p1 = P[1];
  float4 q0 = Q[0], q1 = Q[1];
  float4 b0 = B[0], b1 = B[1];
  float m[8];
  m[0] = fmaxf(p0.x + q0.x + b0.x, 0.f);
  m[1] = fmaxf(p0.y + q0.y + b0.y, 0.f);
  m[2] = fmaxf(p0.z + q0.z + b0.z, 0.f);
  m[3] = fmaxf(p0.w + q0.w + b0.w, 0.f);
  m[4] = fmaxf(p1.x + q1.x + b1.x, 0.f);
  m[5] = fmaxf(p1.y + q1.y + b1.y, 0.f);
  m[6] = fmaxf(p1.z + q1.z + b1.z, 0.f);
  m[7] = fmaxf(p1.w + q1.w + b1.w, 0.f);
  float* dst = agg + (size_t)col * DIM + j;
#pragma unroll
  for (int i = 0; i < 8; ++i) atomicAdd(dst + i, m[i]);
}

// One wave per row: LayerNorm over 256 elements (8 per lane, shfl_xor reduce).
__global__ void __launch_bounds__(256)
layernorm_kernel(const float* __restrict__ h, const float* __restrict__ gamma,
                 const float* __restrict__ beta, float* __restrict__ out) {
  int t = blockIdx.x * blockDim.x + threadIdx.x;
  int rowi = t >> 5;                         // wave-uniform
  if (rowi >= NNODES) return;
  int lane = t & 31;
  int j = lane * 8;
  const float4* hp = (const float4*)(h + (size_t)rowi * DIM + j);
  float4 a = hp[0], b = hp[1];
  float v[8] = {a.x, a.y, a.z, a.w, b.x, b.y, b.z, b.w};
  float s = 0.f, sq = 0.f;
#pragma unroll
  for (int i = 0; i < 8; ++i) { s += v[i]; sq += v[i] * v[i]; }
#pragma unroll
  for (int off = 16; off > 0; off >>= 1) {
    s  += __shfl_xor(s, off, 32);
    sq += __shfl_xor(sq, off, 32);
  }
  const float mu   = s * (1.f / 256.f);
  const float var  = sq * (1.f / 256.f) - mu * mu;
  const float rstd = rsqrtf(var + 1e-5f);
  const float4* g  = (const float4*)(gamma + j);
  const float4* be = (const float4*)(beta + j);
  float4 g0 = g[0], g1 = g[1], e0 = be[0], e1 = be[1];
  float gv[8] = {g0.x, g0.y, g0.z, g0.w, g1.x, g1.y, g1.z, g1.w};
  float ev[8] = {e0.x, e0.y, e0.z, e0.w, e1.x, e1.y, e1.z, e1.w};
  float* op = out + (size_t)rowi * DIM + j;
  float4 o0, o1;
  o0.x = (v[0] - mu) * rstd * gv[0] + ev[0];
  o0.y = (v[1] - mu) * rstd * gv[1] + ev[1];
  o0.z = (v[2] - mu) * rstd * gv[2] + ev[2];
  o0.w = (v[3] - mu) * rstd * gv[3] + ev[3];
  o1.x = (v[4] - mu) * rstd * gv[4] + ev[4];
  o1.y = (v[5] - mu) * rstd * gv[5] + ev[5];
  o1.z = (v[6] - mu) * rstd * gv[6] + ev[6];
  o1.w = (v[7] - mu) * rstd * gv[7] + ev[7];
  ((float4*)op)[0] = o0;
  ((float4*)op)[1] = o1;
}

extern "C" void kernel_launch(void* const* d_in, const int* in_sizes, int n_in,
                              void* d_out, int out_size, void* d_ws, size_t ws_size,
                              hipStream_t stream) {
  (void)in_sizes; (void)n_in; (void)out_size; (void)ws_size;
  const float* x     = (const float*)d_in[0];
  const int*   ei    = (const int*)d_in[1];      // edge_index [2, E]
  const float* Wmsg  = (const float*)d_in[2];    // [512, 256]
  const float* bmsg  = (const float*)d_in[3];    // [256]
  const float* Wupd  = (const float*)d_in[4];    // [512, 256]
  const float* bupd  = (const float*)d_in[5];    // [256]
  const float* gamma = (const float*)d_in[6];    // [256]
  const float* beta  = (const float*)d_in[7];    // [256]
  float* out = (float*)d_out;

  // Workspace layout
  char* ws = (char*)d_ws;
  size_t off = 0;
  auto alloc = [&](size_t bytes) -> void* {
    void* p = ws + off;
    off = (off + bytes + 255) & ~(size_t)255;
    return p;
  };
  float*  PQ     = (float*)alloc((size_t)NNODES * 512 * sizeof(float));   // 20.48 MB
  float*  agg    = (float*)alloc((size_t)NNODES * DIM * sizeof(float));   // 10.24 MB
  bf16_t* WmsgHi = (bf16_t*)alloc((size_t)512 * DIM * sizeof(bf16_t));
  bf16_t* WmsgLo = (bf16_t*)alloc((size_t)512 * DIM * sizeof(bf16_t));
  bf16_t* WupdHi = (bf16_t*)alloc((size_t)512 * DIM * sizeof(bf16_t));
  bf16_t* WupdLo = (bf16_t*)alloc((size_t)512 * DIM * sizeof(bf16_t));
  float*  h      = PQ;  // reuse PQ region after the edge phase

  const size_t kstepElems = 16 * 32 * 16;  // swizzled elements per 32-wide k-step

  // 1) Pre-swizzle weights to hi/lo bf16 B-fragment layout (16 k-steps each).
  swizzle_w_kernel<<<32, 256, 0, stream>>>(Wmsg, 16, WmsgHi, WmsgLo);
  swizzle_w_kernel<<<32, 256, 0, stream>>>(Wupd, 16, WupdHi, WupdLo);

  // 2) Zero the aggregation buffer (atomics target).
  zero_kernel<<<(NNODES * DIM / 4 + 255) / 256, 256, 0, stream>>>(
      (float4*)agg, NNODES * DIM / 4);

  // 3) P = x @ W_msg[0:256,:]   (k-steps 0..7 of the swizzled W_msg)
  gemm_wmma_kernel<<<313, 256, 0, stream>>>(
      x, nullptr, 256, WmsgHi, WmsgLo, PQ, 512, nullptr, 0);
  // 4) Q = x @ W_msg[256:512,:] (k-steps 8..15)
  gemm_wmma_kernel<<<313, 256, 0, stream>>>(
      x, nullptr, 256, WmsgHi + 8 * kstepElems, WmsgLo + 8 * kstepElems,
      PQ + 256, 512, nullptr, 0);

  // 5) Edge phase: msg = relu(P[row] + Q[col] + b_msg); scatter-add into agg[col].
  edge_kernel<<<(NEDGES * 32) / 256, 256, 0, stream>>>(PQ, ei, bmsg, agg);

  // 6) h = relu([x | agg] @ W_upd + b_upd)
  gemm_wmma_kernel<<<313, 256, 0, stream>>>(
      x, agg, 512, WupdHi, WupdLo, h, 256, bupd, 1);

  // 7) LayerNorm(h) * gamma + beta -> out
  layernorm_kernel<<<(NNODES * 32) / 256, 256, 0, stream>>>(h, gamma, beta, out);
}
